// MeshConvNet_3934190043217
// MI455X (gfx1250) — matrix operand order repaired and checked by reading, not compiled, not run on hardware
//
#include <hip/hip_runtime.h>
#include <hip/hip_bf16.h>

// ---------------- constants from the reference ----------------
#define BGR   32
#define N0_   2048
#define EDG   524288           // B*N0*DEG
#define EPS_  1e-5f

typedef __attribute__((ext_vector_type(16))) _Float16 v16h;
typedef __attribute__((ext_vector_type(8)))  float    v8f;
typedef __attribute__((ext_vector_type(4)))  unsigned int u32x4;
typedef __attribute__((ext_vector_type(8)))  int      i32x8;
typedef __attribute__((ext_vector_type(4)))  int      i32x4;

// ---------------- TDM: async 2D tile load (global -> LDS) ----------------
// Builds the Tensor DMA Descriptor (D#) per CDNA5 ISA ch.8:
//  group0: [1:0]=count=1, [63:32]=lds_addr, [120:64]=global_addr, [127:126]=type=2
//  group1: [17:16]=data_size(2 -> 4B), tensor_dim0/1, tile_dim0/1, dim0 stride
__device__ __forceinline__ void tdm_load_2d_f32(unsigned lds_off, const float* gptr,
                                                int cols, int rows, int stride_elems) {
  unsigned long long ga = (unsigned long long)gptr;
  u32x4 g0;
  g0[0] = 1u;                                            // count=1 (valid descriptor)
  g0[1] = lds_off;                                       // lds_addr (bytes)
  g0[2] = (unsigned)(ga & 0xFFFFFFFFull);                // global_addr[31:0]
  g0[3] = (unsigned)((ga >> 32) & 0x1FFFFFFull) | (2u << 30);  // addr[56:32] | type=2
  i32x8 g1;
  g1[0] = (2 << 16);                                     // data_size=4B, wg_mask=0
  g1[1] = (int)(((unsigned)cols & 0xFFFFu) << 16);       // tensor_dim0[15:0] @ bits63:48
  g1[2] = (int)((((unsigned)cols >> 16) & 0xFFFFu)       // tensor_dim0[31:16]
              | (((unsigned)rows & 0xFFFFu) << 16));     // tensor_dim1[15:0]
  g1[3] = (int)((((unsigned)rows >> 16) & 0xFFFFu)       // tensor_dim1[31:16]
              | (((unsigned)cols & 0xFFFFu) << 16));     // tile_dim0
  g1[4] = (int)((unsigned)rows & 0xFFFFu);               // tile_dim1 (tile_dim2=0 -> 2D)
  g1[5] = stride_elems;                                  // tensor_dim0_stride[31:0]
  g1[6] = 0;                                             // stride hi / dim1 stride lo
  g1[7] = 0;
  i32x4 z4 = {0, 0, 0, 0};
#if defined(__clang_major__) && (__clang_major__ >= 23)
  i32x8 z8 = {0, 0, 0, 0, 0, 0, 0, 0};
  __builtin_amdgcn_tensor_load_to_lds(g0, g1, z4, z4, z8, 0);
#else
  __builtin_amdgcn_tensor_load_to_lds(g0, g1, z4, z4, 0);
#endif
}

// ---------------- generic helpers ----------------
__global__ void k_zero(float* __restrict__ p, long long n) {
  long long i = (long long)blockIdx.x * blockDim.x + threadIdx.x;
  if (i < n) p[i] = 0.f;
}

__global__ void k_copy_edges(const int* __restrict__ ei, int* __restrict__ src,
                             int* __restrict__ dst, int* __restrict__ em, int E) {
  int e = blockIdx.x * blockDim.x + threadIdx.x;
  if (e >= E) return;
  src[e] = ei[e];
  dst[e] = ei[E + e];
  em[e]  = 1;
}

// ---------------- WMMA matmul: Y[N,F] = X[N,K] @ W[K,F] ----------------
// fp32 in/out, f16 WMMA with fp32 accumulate. N,F multiples of 16, K multiple of 32.
// Wave 0 TDM-stages the current 32xF chunk of W into LDS (async tensor DMA +
// s_wait_tensorcnt); all 8 waves consume B fragments from LDS. A fragments come
// from global as aligned float2 (the A layout pairs K=2i,2i+1), with a
// global_prefetch of the next K chunk.
__global__ __launch_bounds__(256)
void k_matmul_wmma(const float* __restrict__ X, const float* __restrict__ W,
                   float* __restrict__ Y, int N, int K, int F) {
  __shared__ float sW[32 * 256];                 // one 32-row K-chunk of W (<=32KB)
  int lane   = threadIdx.x & 31;
  int wave   = threadIdx.x >> 5;
  int tilesF = F >> 4;
  int total  = (N >> 4) * tilesF;
  int tile   = blockIdx.x * 8 + wave;
  bool active = tile < total;                    // uniform per wave
  int mt = active ? tile / tilesF : 0;
  int ft = active ? tile % tilesF : 0;
  int m0 = mt << 4, f0 = ft << 4;
  int hl  = lane >> 4;                           // lane half (0/1)
  int l16 = lane & 15;
  int arow = m0 + l16;                           // A: M = lane%16
  int bcol = f0 + l16;                           // B/C: N = lane%16
  unsigned sW_off = (unsigned)(unsigned long long)sW;  // flat addr low 32b = LDS offset
  v8f c = {};
  for (int kk = 0; kk < K; kk += 32) {
    __syncthreads();                             // previous chunk fully consumed
    if (wave == 0) {
      tdm_load_2d_f32(sW_off, W + (long long)kk * F, F, 32, F);
      __builtin_amdgcn_s_wait_tensorcnt(0);
    }
    __syncthreads();                             // chunk visible to all waves
    if (active) {
      const float* Xr = X + (long long)arow * K + kk;
      if (kk + 32 < K) __builtin_prefetch(Xr + 32, 0, 1);
      v16h a, b;
#pragma unroll
      for (int i = 0; i < 8; ++i) {
        // A 16x32 layout: VGPR i<4 -> K = 8*half + 2i,2i+1 ; VGPR 4..7 -> +16
        int ka = (i < 4) ? (8 * hl + 2 * i) : (16 + 8 * hl + 2 * (i - 4));
        float2 p = *(const float2*)(Xr + ka);
        a[2 * i]     = (_Float16)p.x;
        a[2 * i + 1] = (_Float16)p.y;
      }
#pragma unroll
      for (int e = 0; e < 16; ++e) {
        // B 32x16 layout: e<8 -> K = 8*half + e ; e>=8 -> K = 16 + 8*half + (e-8)
        b[e] = (_Float16)sW[((e & 7) + 8 * hl + 16 * (e >> 3)) * F + bcol];
      }
      c = __builtin_amdgcn_wmma_f32_16x16x32_f16(false, a, false, b, (short)0, c,
                                                 false, false);
    }
  }
  if (active) {
#pragma unroll
    for (int r = 0; r < 8; ++r) {
      int row = m0 + r + 8 * hl;                 // C layout: VGPR r -> M = r / r+8
      Y[(long long)row * F + bcol] = c[r];
    }
  }
}

// ---------------- GCNConv pieces ----------------
__global__ void k_deg(const int* __restrict__ dst, const int* __restrict__ em,
                      float* __restrict__ deg, int E) {
  int e = blockIdx.x * blockDim.x + threadIdx.x;
  if (e >= E) return;
  if (em[e]) atomicAdd(&deg[dst[e]], 1.f);
}

__global__ void k_dis(float* __restrict__ d, int n) {   // in-place deg -> rsqrt(deg+1)
  int v = blockIdx.x * blockDim.x + threadIdx.x;
  if (v >= n) return;
  d[v] = rsqrtf(d[v] + 1.f);
}

// out[v,f] = dis[v]^2 * xw[v,f] + bias[f]   (self-loop term + bias, init for edge agg)
__global__ void k_selfloop(const float* __restrict__ xw, const float* __restrict__ dis,
                           const float* __restrict__ bias, float* __restrict__ out,
                           long long nf, int F) {
  long long i = (long long)blockIdx.x * blockDim.x + threadIdx.x;
  if (i >= nf) return;
  int v = (int)(i / F), f = (int)(i % F);
  float d = dis[v];
  out[i] = d * d * xw[i] + bias[f];
}

// out[dst] += dis[src]*dis[dst]*xw[src]  (4 features per thread, b128 gather)
__global__ void k_edge_agg(const int* __restrict__ src, const int* __restrict__ dst,
                           const int* __restrict__ em, const float* __restrict__ dis,
                           const float* __restrict__ xw, float* __restrict__ out,
                           int E, int F) {
  int fq = F >> 2;
  long long t = (long long)blockIdx.x * blockDim.x + threadIdx.x;
  if (t >= (long long)E * fq) return;
  int e  = (int)(t / fq);
  if (!em[e]) return;
  int f4 = ((int)(t % fq)) << 2;
  int s = src[e], d = dst[e];
  float coef = dis[s] * dis[d];
  float4 xs = *(const float4*)(xw + (long long)s * F + f4);
  float* od = out + (long long)d * F + f4;
  atomicAdd(od + 0, coef * xs.x);
  atomicAdd(od + 1, coef * xs.y);
  atomicAdd(od + 2, coef * xs.z);
  atomicAdd(od + 3, coef * xs.w);
}

// ---------------- ReLU + BatchNorm ----------------
__global__ __launch_bounds__(256)
void k_relu_bnstats(float* __restrict__ x, float* __restrict__ stats,
                    long long nf, int F) {
  __shared__ float s_sum[128], s_sq[128];
  for (int f = threadIdx.x; f < F; f += blockDim.x) { s_sum[f] = 0.f; s_sq[f] = 0.f; }
  __syncthreads();
  long long base = (long long)blockIdx.x * blockDim.x * 16;
  for (int it = 0; it < 16; ++it) {
    long long i = base + (long long)it * blockDim.x + threadIdx.x;
    if (i < nf) {
      float v = x[i];
      v = v > 0.f ? v : 0.f;
      x[i] = v;
      int f = (int)(i % F);
      atomicAdd(&s_sum[f], v);
      atomicAdd(&s_sq[f], v * v);
    }
  }
  __syncthreads();
  for (int f = threadIdx.x; f < F; f += blockDim.x) {
    atomicAdd(&stats[f], s_sum[f]);
    atomicAdd(&stats[F + f], s_sq[f]);
  }
}

__global__ void k_bn_apply(float* __restrict__ x, const float* __restrict__ stats,
                           const float* __restrict__ g, const float* __restrict__ b,
                           long long nf, int n, int F) {
  long long i = (long long)blockIdx.x * blockDim.x + threadIdx.x;
  if (i >= nf) return;
  int f = (int)(i % F);
  float inv_n = 1.f / (float)n;
  float mu  = stats[f] * inv_n;
  float var = stats[F + f] * inv_n - mu * mu;
  x[i] = (x[i] - mu) * rsqrtf(var + EPS_) * g[f] + b[f];
}

// ---------------- SAGPool ----------------
// score[v] = x[v].Wroot + pb ; srel[v] = x[v].Wrel  (edge kernel adds srel[src] to score[dst])
__global__ void k_score_root(const float* __restrict__ x, const float* __restrict__ Wroot,
                             const float* __restrict__ Wrel, const float* __restrict__ pb,
                             float* __restrict__ score, float* __restrict__ srel,
                             int n, int F) {
  int v = blockIdx.x * blockDim.x + threadIdx.x;
  if (v >= n) return;
  const float* xv = x + (long long)v * F;
  float sr = 0.f, sl = 0.f;
  for (int f = 0; f < F; ++f) { sr += xv[f] * Wroot[f]; sl += xv[f] * Wrel[f]; }
  score[v] = sr + pb[0];
  srel[v]  = sl;
}

__global__ void k_score_edge(const int* __restrict__ src, const int* __restrict__ dst,
                             const int* __restrict__ em, const float* __restrict__ srel,
                             float* __restrict__ score, int E) {
  int e = blockIdx.x * blockDim.x + threadIdx.x;
  if (e >= E) return;
  if (em[e]) atomicAdd(&score[dst[e]], srel[src[e]]);
}

// exact top_k semantics: rank = #(score[j] > score[i]) + #(ties with j < i)
__global__ __launch_bounds__(256)
void k_rank(const float* __restrict__ score, int* __restrict__ rank, int npg) {
  __shared__ float s[2048];
  int g = blockIdx.x;
  const float* sg = score + (long long)g * npg;
  for (int i = threadIdx.x; i < npg; i += blockDim.x) s[i] = sg[i];
  __syncthreads();
  for (int i = threadIdx.x; i < npg; i += blockDim.x) {
    float si = s[i];
    int cnt = 0;
    for (int j = 0; j < npg; ++j) {
      float sj = s[j];
      cnt += (sj > si) || (sj == si && j < i);
    }
    rank[(long long)g * npg + i] = cnt;
  }
}

__global__ void k_pool_gather(const float* __restrict__ xin, const float* __restrict__ score,
                              const int* __restrict__ rank, float* __restrict__ xout,
                              int* __restrict__ new_id, int npg, int k, int F, int n) {
  int v = blockIdx.x * blockDim.x + threadIdx.x;
  if (v >= n) return;
  int g = v / npg;
  int r = rank[v];
  if (r < k) {
    int nid = g * k + r;
    new_id[v] = nid;
    float t = tanhf(score[v]);
    const float* xi = xin + (long long)v * F;
    float* xo = xout + (long long)nid * F;
    for (int f = 0; f < F; ++f) xo[f] = xi[f] * t;
  } else {
    new_id[v] = -1;
  }
}

__global__ void k_relabel(int* __restrict__ src, int* __restrict__ dst,
                          int* __restrict__ em, const int* __restrict__ new_id, int E) {
  int e = blockIdx.x * blockDim.x + threadIdx.x;
  if (e >= E) return;
  int ns = new_id[src[e]], nd = new_id[dst[e]];
  int m = em[e] && (ns >= 0) && (nd >= 0);
  src[e] = m ? ns : 0;
  dst[e] = m ? nd : 0;
  em[e]  = m;
}

// ---------------- head ----------------
__global__ void k_mean_pool(const float* __restrict__ x, float* __restrict__ emb,
                            int npg, int F) {
  int t = blockIdx.x * blockDim.x + threadIdx.x;   // B*F
  if (t >= BGR * F) return;
  int b = t / F, f = t % F;
  const float* xb = x + (long long)b * npg * F + f;
  float s = 0.f;
  for (int i = 0; i < npg; ++i) s += xb[(long long)i * F];
  emb[t] = s / (float)npg;
}

// out[i,c] = sum_j relu(hraw[i,j]+b1[j]) * W2[j,c] + b2[c]
__global__ void k_fc2(const float* __restrict__ hraw, const float* __restrict__ b1,
                      const float* __restrict__ W2, const float* __restrict__ b2,
                      float* __restrict__ out) {
  int t = blockIdx.x * blockDim.x + threadIdx.x;   // 32*40
  if (t >= BGR * 40) return;
  int i = t / 40, c = t % 40;
  float s = b2[c];
  const float* h = hraw + i * 256;
  for (int j = 0; j < 256; ++j) {
    float v = h[j] + b1[j];
    v = v > 0.f ? v : 0.f;
    s += v * W2[j * 40 + c];
  }
  out[t] = s;
}

// ---------------- host orchestration ----------------
extern "C" void kernel_launch(void* const* d_in, const int* in_sizes, int n_in,
                              void* d_out, int out_size, void* d_ws, size_t ws_size,
                              hipStream_t stream) {
  (void)in_sizes; (void)n_in; (void)out_size; (void)ws_size;

  static const int NPGc[4]  = {2048, 1639, 1312, 1050};
  static const int DIMSc[4] = {32, 64, 128, 128};

  const float* x0   = (const float*)d_in[0];
  const int*   eidx = (const int*)d_in[1];
  const float* convW[3] = {(const float*)d_in[3],  (const float*)d_in[10], (const float*)d_in[17]};
  const float* convB[3] = {(const float*)d_in[4],  (const float*)d_in[11], (const float*)d_in[18]};
  const float* normG[3] = {(const float*)d_in[5],  (const float*)d_in[12], (const float*)d_in[19]};
  const float* normB[3] = {(const float*)d_in[6],  (const float*)d_in[13], (const float*)d_in[20]};
  const float* Wrel[3]  = {(const float*)d_in[7],  (const float*)d_in[14], (const float*)d_in[21]};
  const float* Wroot[3] = {(const float*)d_in[8],  (const float*)d_in[15], (const float*)d_in[22]};
  const float* poolB[3] = {(const float*)d_in[9],  (const float*)d_in[16], (const float*)d_in[23]};
  const float* fc1W = (const float*)d_in[24];
  const float* fc1b = (const float*)d_in[25];
  const float* fc2W = (const float*)d_in[26];
  const float* fc2b = (const float*)d_in[27];

  // workspace carve-up (floats).  total ~83 MB
  float* ws = (float*)d_ws;
  size_t o = 0;
  float* buf0  = ws + o; o += 5374976;   // pooled x (max 41984*128)
  float* buf1  = ws + o; o += 6713600;   // conv output (max 52448*128)
  float* xwb   = ws + o; o += 6713600;   // x@W      (max 52448*128)
  float* dis   = ws + o; o += 65536;     // deg -> rsqrt(deg+1)
  float* score = ws + o; o += 65536;
  float* srel  = ws + o; o += 65536;
  float* stats = ws + o; o += 512;       // [sum(F) | sumsq(F)]
  float* hraw  = ws + o; o += 8192;      // fc1 raw output 32x256
  int* rnk   = (int*)(ws + o); o += 65536;
  int* newid = (int*)(ws + o); o += 65536;
  int* srcb  = (int*)(ws + o); o += EDG;
  int* dstb  = (int*)(ws + o); o += EDG;
  int* emsk  = (int*)(ws + o); o += EDG;

  const int T = 256;
  const int EB = (EDG + T - 1) / T;

  k_copy_edges<<<EB, T, 0, stream>>>(eidx, srcb, dstb, emsk, EDG);

  const float* xcur = x0;
  for (int i = 0; i < 3; ++i) {
    int n = BGR * NPGc[i];
    int K = DIMSc[i], F = DIMSc[i + 1];
    long long nf = (long long)n * F;

    // 1. xw = x @ W   (WMMA + TDM-staged weights)
    int tiles = (n >> 4) * (F >> 4);
    k_matmul_wmma<<<(tiles + 7) / 8, T, 0, stream>>>(xcur, convW[i], xwb, n, K, F);

    // 2. degree + rsqrt
    k_zero<<<(n + T - 1) / T, T, 0, stream>>>(dis, n);
    k_deg<<<EB, T, 0, stream>>>(dstb, emsk, dis, EDG);
    k_dis<<<(n + T - 1) / T, T, 0, stream>>>(dis, n);

    // 3. self-loop term + bias init, then edge aggregation into buf1
    k_selfloop<<<(int)((nf + T - 1) / T), T, 0, stream>>>(xwb, dis, convB[i], buf1, nf, F);
    long long et = (long long)EDG * (F >> 2);
    k_edge_agg<<<(int)((et + T - 1) / T), T, 0, stream>>>(srcb, dstb, emsk, dis, xwb, buf1, EDG, F);

    // 4. ReLU + BatchNorm (training mode, biased variance)
    k_zero<<<1, T, 0, stream>>>(stats, 2 * F);
    k_relu_bnstats<<<(int)((nf + 4095) / 4096), T, 0, stream>>>(buf1, stats, nf, F);
    k_bn_apply<<<(int)((nf + T - 1) / T), T, 0, stream>>>(buf1, stats, normG[i], normB[i], nf, n, F);

    // 5. SAGPool scores: GraphConv(F,1)
    k_score_root<<<(n + T - 1) / T, T, 0, stream>>>(buf1, Wroot[i], Wrel[i], poolB[i], score, srel, n, F);
    k_score_edge<<<EB, T, 0, stream>>>(srcb, dstb, emsk, srel, score, EDG);

    // 6. exact per-graph top-k via rank; gather + tanh scale; relabel edges
    k_rank<<<BGR, T, 0, stream>>>(score, rnk, NPGc[i]);
    k_pool_gather<<<(n + T - 1) / T, T, 0, stream>>>(buf1, score, rnk, buf0, newid,
                                                     NPGc[i], NPGc[i + 1], F, n);
    k_relabel<<<EB, T, 0, stream>>>(srcb, dstb, emsk, newid, EDG);

    xcur = buf0;
  }

  // global mean pool -> emb goes straight into d_out[1280:]
  float* outp = (float*)d_out;
  float* embp = outp + BGR * 40;
  k_mean_pool<<<(BGR * 128 + T - 1) / T, T, 0, stream>>>(xcur, embp, NPGc[3], 128);

  // fc1 (WMMA, 32x128 @ 128x256), fc2 fused with bias+relu of fc1
  int t1 = (32 >> 4) * (256 >> 4);
  k_matmul_wmma<<<(t1 + 7) / 8, T, 0, stream>>>(embp, fc1W, hraw, 32, 128, 256);
  k_fc2<<<(BGR * 40 + T - 1) / T, T, 0, stream>>>(hraw, fc1b, fc2W, fc2b, outp);
}
